// TriangleMultiplicationTorch_62182536511496
// MI455X (gfx1250) — compile-verified, matrix operand order checked by reading, not run on hardware
//
#include <hip/hip_runtime.h>
#include <hip/hip_bf16.h>

typedef __attribute__((ext_vector_type(16))) _Float16 v16h;
typedef __attribute__((ext_vector_type(8)))  _Float16 h8;
typedef __attribute__((ext_vector_type(8)))  float    v8f;
typedef __attribute__((ext_vector_type(4)))  float    f4;
typedef unsigned int u32x4 __attribute__((ext_vector_type(4)));
typedef int          i32x8 __attribute__((ext_vector_type(8)));
typedef int          i32x4 __attribute__((ext_vector_type(4)));

#ifndef __has_builtin
#define __has_builtin(x) 0
#endif
#if __has_builtin(__builtin_amdgcn_tensor_load_to_lds) && \
    __has_builtin(__builtin_amdgcn_s_wait_tensorcnt)
#define HAVE_TDM 1
#warning "PROBE: CDNA5 TDM path ENABLED (tensor_load_to_lds emitted in k2)"
#else
#define HAVE_TDM 0
#warning "PROBE: CDNA5 TDM builtin NOT available - manual LDS staging fallback in k2"
#endif

constexpr int NN  = 512;
constexpr int CC  = 128;
constexpr int NSQ = NN * NN;          // 262144 positions

// ---- workspace layout (bytes) ----
constexpr size_t OFF_W   = 0;                          // 640*128 f16 (Wp|Wg|Wgl rows)
constexpr size_t OFF_WO  = 256u * 1024u;               // 128*128 f16
constexpr size_t OFF_A   = 1u * 1024u * 1024u;         // a_t  f16 (C,N,N) 64MB
constexpr size_t OFF_B   = OFF_A + (size_t)64 * 1024 * 1024;   // b_t  f16 64MB
constexpr size_t OFF_GL  = OFF_B + (size_t)64 * 1024 * 1024;   // gl   f16 (N,N,C) 64MB
constexpr size_t OFF_TRI = OFF_GL + (size_t)64 * 1024 * 1024;  // tri  f32 (C,N,N) 128MB

__device__ __forceinline__ v8f wmma16(v16h a, v16h b, v8f c) {
  return __builtin_amdgcn_wmma_f32_16x16x32_f16(false, a, false, b, (short)0, c,
                                                false, false);
}

// Load one 16x32 operand tile from a row-major f16 buffer (stride in f16 units).
// Per ISA 16-bit A/B layout: lane L<16 -> row+L, K = {0..7,16..23};
// lanes 16..31 -> same rows, K = {8..15,24..31}. Two 16B chunks per lane.
__device__ __forceinline__ v16h load_tile(const _Float16* base, int row0, int k0,
                                          int stride) {
  int lane = threadIdx.x & 31;
  int r = row0 + (lane & 15);
  int kh = (lane >> 4) << 3;            // 0 or 8
  const _Float16* p = base + (size_t)r * stride + k0;
  union { v16h v; h8 h[2]; } u;
  u.h[0] = *(const h8*)(p + kh);
  u.h[1] = *(const h8*)(p + 16 + kh);
  return u.v;
}

// Store a 16x16 f32 D tile into row-major f32 (stride in floats).
// D layout: VGPR d, lanes 0-15: row=row0+d, col=lane; lanes 16-31: row=row0+8+d.
__device__ __forceinline__ void store_tile_f32(float* base, int row0, int col0,
                                               int stride, v8f acc) {
  int lane = threadIdx.x & 31;
  int col = col0 + (lane & 15);
  int r = row0 + ((lane >> 4) << 3);
#pragma unroll
  for (int d = 0; d < 8; ++d) base[(size_t)(r + d) * stride + col] = acc[d];
}

// Fast sigmoid: v_exp + v_rcp instead of IEEE-exact division.
__device__ __forceinline__ float sigmoidf(float x) {
  return __builtin_amdgcn_rcpf(1.0f + __expf(-x));
}

#if HAVE_TDM
// Issue a TDM 2D tile load: tile_rows x 32 f16 from a row-major (.,512) f16
// tensor into LDS with +16B padding per 64B row (matches [rows][40] f16 layout).
// D# bit packing per CDNA5 ISA.
__device__ __forceinline__ void tdm_load_2d(unsigned lds_off, const void* gsrc,
                                            int tile_rows) {
  unsigned long long ga = (unsigned long long)(uintptr_t)gsrc;
  u32x4 g0 = { 1u,                                  // count=1, is_restore=0, gather=0
               lds_off,                              // lds_addr (bytes)
               (unsigned)ga,                         // global_addr[31:0]
               (unsigned)((ga >> 32) & 0x1FFFFFFu) | 0x80000000u }; // addr[56:32]|type=2
  i32x8 g1;
  g1[0] = (int)0x06D10000u;   // data_size=2B(1), pad_enable, pad_interval=16DW(3),
                              // pad_amount=4DW(3), wg_mask=0
  g1[1] = (int)0x02000000u;   // atomic_barrier=0 | tensor_dim0(512) low16 << 16
  g1[2] = (int)0x02000000u;   // tensor_dim0 hi16=0 | tensor_dim1(512) low16 << 16
  g1[3] = (int)(32u << 16);   // tensor_dim1 hi16=0 | tile_dim0 = 32 elements
  g1[4] = tile_rows;          // tile_dim1 = rows, tile_dim2 = 0
  g1[5] = 512;                // tensor_dim0_stride low32 (elements)
  g1[6] = 0;
  g1[7] = 0;                  // tensor_dim1_stride unused (2D tile)
  i32x4 z4 = {0, 0, 0, 0};
#if __clang_major__ >= 23
  i32x8 z8 = {0, 0, 0, 0, 0, 0, 0, 0};
  __builtin_amdgcn_tensor_load_to_lds(g0, g1, z4, z4, z8, 0);
#else
  __builtin_amdgcn_tensor_load_to_lds(g0, g1, z4, z4, 0);
#endif
}
#endif

// ---------------- kernel 0: weights -> f16 workspace ----------------
__global__ void k0_convert(const float* __restrict__ Wp, const float* __restrict__ Wg,
                           const float* __restrict__ Wgl, const float* __restrict__ Wo,
                           _Float16* __restrict__ wsW, _Float16* __restrict__ wsWo) {
  int i = blockIdx.x * blockDim.x + threadIdx.x;
  if (i < 256 * 128)        wsW[i] = (_Float16)Wp[i];
  else if (i < 512 * 128)   wsW[i] = (_Float16)Wg[i - 256 * 128];
  else if (i < 640 * 128)   wsW[i] = (_Float16)Wgl[i - 512 * 128];
  if (i < 128 * 128)        wsWo[i] = (_Float16)Wo[i];
}

// ---------------- kernel 1: LN1 + Wp/Wg/Wgl GEMM + gate/deinterleave ----------------
__global__ void k1_ln_proj(const float* __restrict__ pair, const float* __restrict__ mask,
                           const float* __restrict__ ln1w, const float* __restrict__ ln1b,
                           const _Float16* __restrict__ wsW,
                           _Float16* __restrict__ a_t, _Float16* __restrict__ b_t,
                           _Float16* __restrict__ gl) {
  __shared__ float    P[64][128];       // 32 KB raw pair rows
  __shared__ _Float16 Xh[64][136];      // normalized f16, 16B-aligned rows

  const int tid = threadIdx.x;
  const int r0 = blockIdx.x * 64;

  // phase 1: stage 64 rows of pair (float4)
  const f4* pp = (const f4*)(pair + (size_t)r0 * CC);
#pragma unroll
  for (int it = 0; it < 8; ++it) {
    int idx = it * 256 + tid;           // 2048 float4 total
    f4 v = pp[idx];
    int row = idx >> 5, c4 = idx & 31;
    *(f4*)&P[row][c4 * 4] = v;
  }
  __syncthreads();

  // phase 2: layernorm (4 lanes per row, shfl reduce)
  {
    int rr = tid >> 2, part = tid & 3, pc = part * 32;
    float s = 0.f, s2 = 0.f;
#pragma unroll
    for (int c = 0; c < 32; ++c) { float v = P[rr][pc + c]; s += v; s2 += v * v; }
    s  += __shfl_xor(s, 1, 32);  s  += __shfl_xor(s, 2, 32);
    s2 += __shfl_xor(s2, 1, 32); s2 += __shfl_xor(s2, 2, 32);
    float mu = s * (1.0f / 128.0f);
    float var = s2 * (1.0f / 128.0f) - mu * mu;
    float rs = rsqrtf(var + 1e-5f);
#pragma unroll
    for (int c = 0; c < 32; ++c) {
      float x = (P[rr][pc + c] - mu) * rs * ln1w[pc + c] + ln1b[pc + c];
      Xh[rr][pc + c] = (_Float16)x;
    }
  }
  __syncthreads();

  // phase 3: WMMA GEMM. Each wave owns one M-tile (m = wave&3) and one output
  // parity (wave>>2); its 4 K-step A operands are loaded from LDS exactly once.
  const int wave = tid >> 5, lane = tid & 31;
  const int m0 = (wave & 3) * 16;
  const _Float16* Wb  = wsW;                 // proj rows [0,256)
  const _Float16* Wgb = wsW + 256 * 128;     // gate rows
  const _Float16* Wlb = wsW + 512 * 128;     // gl rows

  v16h Av[4];
#pragma unroll
  for (int ks = 0; ks < 4; ++ks) Av[ks] = load_tile(&Xh[0][0], m0, ks * 32, 136);

  const int rbase = r0 + m0 + ((lane >> 4) << 3);   // 8 consecutive positions

  for (int t = 0; t < 12; ++t) {
    int tt = (wave >> 2) + 2 * t;            // 0..23, split by parity across waves
    if (tt < 16) {                           // paired proj/gate output tile
      v8f accp = {}, accg = {};
#pragma unroll
      for (int ks = 0; ks < 4; ++ks) {
        int k0 = ks * 32;
        v16h bp = load_tile(Wb,  tt * 16, k0, 128);
        v16h bg = load_tile(Wgb, tt * 16, k0, 128);
        accp = wmma16(Av[ks], bp, accp);
        accg = wmma16(Av[ks], bg, accg);
      }
      int col = tt * 16 + (lane & 15);       // proj column 0..255
      int ch = col >> 1;
      _Float16* dst = (((col & 1) == 0) ? a_t : b_t) + (size_t)ch * NSQ + rbase;
      union { h8 v; _Float16 e[8]; } o;
#pragma unroll
      for (int d = 0; d < 8; ++d) {
        float g = sigmoidf(accg[d]);
        o.e[d] = (_Float16)(accp[d] * mask[rbase + d] * g);
      }
      *(h8*)dst = o.v;                       // 8 consecutive positions, 16B store
    } else {                                 // Wgl gate tile
      int og = (tt - 16) * 16;
      v8f acc = {};
#pragma unroll
      for (int ks = 0; ks < 4; ++ks) {
        v16h b = load_tile(Wlb, og, ks * 32, 128);
        acc = wmma16(Av[ks], b, acc);
      }
      int col = og + (lane & 15);
#pragma unroll
      for (int d = 0; d < 8; ++d)
        gl[(size_t)(rbase + d) * CC + col] = (_Float16)sigmoidf(acc[d]);
    }
  }
}

// ---------------- kernel 2: tri[c] = A_c * B_c^T (128 x 512^3 GEMMs) ----------------
// Block = 256x128 output tile of one channel. K=512 in 16 steps of 32.
// Staging: TDM tensor_load_to_lds (padded rows), double buffered.
// Waves: (wm 0..3, wn 0..1) -> 64x64 sub-tile = 16 WMMA per K-step.
__global__ void k2_tri(const _Float16* __restrict__ a_t, const _Float16* __restrict__ b_t,
                       float* __restrict__ tri) {
  __shared__ _Float16 As[2][256][40];   // 40 f16 = 80B rows (64B data + 16B pad)
  __shared__ _Float16 Bs[2][128][40];

  const int tid = threadIdx.x;
  const int c  = blockIdx.x >> 3;
  const int i0 = ((blockIdx.x >> 2) & 1) * 256;
  const int j0 = (blockIdx.x & 3) * 128;

  const _Float16* Ab = a_t + (size_t)c * NSQ;
  const _Float16* Bb = b_t + (size_t)c * NSQ;

  const int wave = tid >> 5;
  const int wm = wave & 3;              // M quarter: rows wm*64 + {0..3}*16
  const int wn = wave >> 2;             // N half:    cols wn*64 + {0..3}*16

  v8f acc[4][4];
#pragma unroll
  for (int p = 0; p < 4; ++p)
#pragma unroll
    for (int q = 0; q < 4; ++q) acc[p][q] = (v8f){};

#if HAVE_TDM
  const unsigned ldsA0 = (unsigned)(uintptr_t)&As[0][0][0];
  const unsigned ldsA1 = (unsigned)(uintptr_t)&As[1][0][0];
  const unsigned ldsB0 = (unsigned)(uintptr_t)&Bs[0][0][0];
  const unsigned ldsB1 = (unsigned)(uintptr_t)&Bs[1][0][0];
  // preload stage 0 via TDM (wave 0 -> A slab, wave 1 -> B slab)
  if (wave == 0) {
    tdm_load_2d(ldsA0, Ab + (size_t)i0 * NN, 256);
    __builtin_amdgcn_s_wait_tensorcnt(0);
  } else if (wave == 1) {
    tdm_load_2d(ldsB0, Bb + (size_t)j0 * NN, 128);
    __builtin_amdgcn_s_wait_tensorcnt(0);
  }
#else
  {
    // manual fallback staging for stage 0
    const h8* sa = (const h8*)(Ab + (size_t)(i0 + tid) * NN);
#pragma unroll
    for (int x = 0; x < 4; ++x) *(h8*)&As[0][tid][x * 8] = sa[x];
    int brow = tid & 127, bh = (tid >> 7) * 16;
    const h8* sb = (const h8*)(Bb + (size_t)(j0 + brow) * NN + bh);
    *(h8*)&Bs[0][brow][bh] = sb[0]; *(h8*)&Bs[0][brow][bh + 8] = sb[1];
  }
#endif
  __syncthreads();

  for (int ks = 0; ks < 16; ++ks) {
    int buf = ks & 1;
    if (ks + 1 < 16) {                  // stage next K-slab into other buffer
      int k0 = (ks + 1) * 32;
#if HAVE_TDM
      if (wave == 0) tdm_load_2d(buf ? ldsA0 : ldsA1, Ab + (size_t)i0 * NN + k0, 256);
      else if (wave == 1) tdm_load_2d(buf ? ldsB0 : ldsB1, Bb + (size_t)j0 * NN + k0, 128);
#else
      int nb = buf ^ 1;
      const h8* sa = (const h8*)(Ab + (size_t)(i0 + tid) * NN + k0);
#pragma unroll
      for (int x = 0; x < 4; ++x) *(h8*)&As[nb][tid][x * 8] = sa[x];
      int brow = tid & 127, bh = (tid >> 7) * 16;
      const h8* sb = (const h8*)(Bb + (size_t)(j0 + brow) * NN + k0 + bh);
      *(h8*)&Bs[nb][brow][bh] = sb[0]; *(h8*)&Bs[nb][brow][bh + 8] = sb[1];
#endif
    }
    v16h Av[4];
#pragma unroll
    for (int p = 0; p < 4; ++p)
      Av[p] = load_tile(&As[buf][0][0], wm * 64 + p * 16, 0, 40);
#pragma unroll
    for (int q = 0; q < 4; ++q) {
      v16h bq = load_tile(&Bs[buf][0][0], wn * 64 + q * 16, 0, 40);
#pragma unroll
      for (int p = 0; p < 4; ++p) acc[p][q] = wmma16(Av[p], bq, acc[p][q]);
    }
#if HAVE_TDM
    if (wave < 2) __builtin_amdgcn_s_wait_tensorcnt(0);
#endif
    __syncthreads();
  }

  float* T = tri + (size_t)c * NSQ;
#pragma unroll
  for (int p = 0; p < 4; ++p)
#pragma unroll
    for (int q = 0; q < 4; ++q)
      store_tile_f32(T, i0 + wm * 64 + p * 16, j0 + wn * 64 + q * 16, NN, acc[p][q]);
}

// ---------------- kernel 3: LN2 + Wo GEMM + gate + residual ----------------
__global__ void k3_out(const float* __restrict__ tri, const _Float16* __restrict__ wsWo,
                       const float* __restrict__ ln2w, const float* __restrict__ ln2b,
                       const _Float16* __restrict__ gl, const float* __restrict__ pair,
                       float* __restrict__ out) {
  __shared__ float    X[64][129];
  __shared__ _Float16 Xh[64][136];
  __shared__ _Float16 Woh[128][136];

  const int tid = threadIdx.x;
  const int i  = blockIdx.x >> 3;
  const int j0 = (blockIdx.x & 7) * 64;

  // stage Wo (f16, 16384 elems) into LDS row-major [n][k]
#pragma unroll
  for (int it = 0; it < 8; ++it) {
    int ch = it * 256 + tid;            // 2048 h8 chunks
    int row = ch >> 4, c8 = (ch & 15) * 8;
    *(h8*)&Woh[row][c8] = ((const h8*)wsWo)[ch];
  }
  // gather tri[c][i][j0+jj] -> X[jj][c] (coalesced over jj; tri is L2-resident)
#pragma unroll
  for (int it = 0; it < 32; ++it) {
    int e = it * 256 + tid;             // 8192 elems
    int cch = e >> 6, jj = e & 63;
    X[jj][cch] = tri[(size_t)cch * NSQ + (size_t)i * NN + j0 + jj];
  }
  __syncthreads();

  // LN2 per position
  {
    int rr = tid >> 2, part = tid & 3, pc = part * 32;
    float s = 0.f, s2 = 0.f;
#pragma unroll
    for (int cdx = 0; cdx < 32; ++cdx) { float v = X[rr][pc + cdx]; s += v; s2 += v * v; }
    s  += __shfl_xor(s, 1, 32);  s  += __shfl_xor(s, 2, 32);
    s2 += __shfl_xor(s2, 1, 32); s2 += __shfl_xor(s2, 2, 32);
    float mu = s * (1.0f / 128.0f);
    float var = s2 * (1.0f / 128.0f) - mu * mu;
    float rs = rsqrtf(var + 1e-5f);
#pragma unroll
    for (int cdx = 0; cdx < 32; ++cdx) {
      float x = (X[rr][pc + cdx] - mu) * rs * ln2w[pc + cdx] + ln2b[pc + cdx];
      Xh[rr][pc + cdx] = (_Float16)x;
    }
  }
  __syncthreads();

  // GEMM Y(64x128) = Xh * Wo^T; wave w: m-tile = w&3, n-tiles = (w>>2)*4 + {0..3}
  const int wave = tid >> 5, lane = tid & 31;
  const int m0 = (wave & 3) * 16;
  const int nb = (wave >> 2) * 64;

  v8f acc[4];
#pragma unroll
  for (int t = 0; t < 4; ++t) acc[t] = (v8f){};
#pragma unroll
  for (int ks = 0; ks < 4; ++ks) {
    int k0 = ks * 32;
    v16h a = load_tile(&Xh[0][0], m0, k0, 136);
#pragma unroll
    for (int t = 0; t < 4; ++t) {
      v16h b = load_tile(&Woh[0][0], nb + t * 16, k0, 136);
      acc[t] = wmma16(a, b, acc[t]);
    }
  }

  // epilogue: out = pair + y * gl
#pragma unroll
  for (int t = 0; t < 4; ++t) {
    int col = nb + t * 16 + (lane & 15);
    int jj = m0 + ((lane >> 4) << 3);
#pragma unroll
    for (int d = 0; d < 8; ++d) {
      size_t idx = ((size_t)i * NN + j0 + jj + d) * CC + col;
      float g = (float)gl[idx];
      out[idx] = pair[idx] + acc[t][d] * g;
    }
  }
}

extern "C" void kernel_launch(void* const* d_in, const int* in_sizes, int n_in,
                              void* d_out, int out_size, void* d_ws, size_t ws_size,
                              hipStream_t stream) {
  const float* pair = (const float*)d_in[0];
  const float* mask = (const float*)d_in[1];
  const float* ln1w = (const float*)d_in[2];
  const float* ln1b = (const float*)d_in[3];
  const float* Wp   = (const float*)d_in[4];
  const float* Wg   = (const float*)d_in[5];
  const float* ln2w = (const float*)d_in[6];
  const float* ln2b = (const float*)d_in[7];
  const float* Wo   = (const float*)d_in[8];
  const float* Wgl  = (const float*)d_in[9];
  float* out = (float*)d_out;

  char* ws = (char*)d_ws;
  _Float16* wsW  = (_Float16*)(ws + OFF_W);
  _Float16* wsWo = (_Float16*)(ws + OFF_WO);
  _Float16* a_t  = (_Float16*)(ws + OFF_A);
  _Float16* b_t  = (_Float16*)(ws + OFF_B);
  _Float16* gl   = (_Float16*)(ws + OFF_GL);
  float*    tri  = (float*)   (ws + OFF_TRI);

  k0_convert<<<320, 256, 0, stream>>>(Wp, Wg, Wgl, Wo, wsW, wsWo);
  k1_ln_proj<<<NSQ / 64, 256, 0, stream>>>(pair, mask, ln1w, ln1b, wsW, a_t, b_t, gl);
  k2_tri<<<CC * 8, 256, 0, stream>>>(a_t, b_t, tri);
  k3_out<<<NN * (NN / 64), 256, 0, stream>>>(tri, wsWo, ln2w, ln2b, gl, pair, out);
}
